// CasualSelfAttention_5506148073688
// MI455X (gfx1250) — matrix operand order repaired and checked by reading, not compiled, not run on hardware
//
#include <hip/hip_runtime.h>

typedef __attribute__((ext_vector_type(16))) _Float16 v16h;
typedef __attribute__((ext_vector_type(8)))  _Float16 v8h;
typedef __attribute__((ext_vector_type(8)))  float    v8f;
typedef __attribute__((ext_vector_type(4)))  unsigned int v4u;
typedef __attribute__((ext_vector_type(8)))  unsigned int v8u;

#define B_   2
#define T_   2048
#define C_   1024
#define H_   16
#define D_   64
#define M_   (B_ * T_)    // 4096 rows
#define N3C  (3 * C_)     // 3072
#define BH_  (B_ * H_)    // 32

// ---------- WMMA helpers (CDNA5 16x16x32 f16, wave32) ----------

__device__ __forceinline__ v8f wmma16(v16h a, v16h b, v8f c) {
  // (neg_a, A, neg_b, B, c_mod, C, reuse_a, reuse_b)
  return __builtin_amdgcn_wmma_f32_16x16x32_f16(false, a, false, b, (short)0, c,
                                                false, false);
}

// A fragment 16(M) x 32(K) from row-major f16 [.. x ld].
// ISA layout: lane<16 -> row=lane, K={0..7,16..23}; lane>=16 -> row=lane-16, K={8..15,24..31}.
__device__ __forceinline__ v16h load_frag_a(const _Float16* __restrict__ m,
                                            int row0, int k0, int ld, int lane) {
  int row = row0 + (lane & 15);
  int kb  = k0 + ((lane & 16) ? 8 : 0);
  const _Float16* p = m + (size_t)row * ld + kb;
  v8h lo = *(const v8h*)p;
  v8h hi = *(const v8h*)(p + 16);
  v16h a;
#pragma unroll
  for (int i = 0; i < 8; ++i) { a[i] = lo[i]; a[i + 8] = hi[i]; }
  return a;
}

// B fragment 32(K) x 16(N) from row-major f16 [K x ld] (row = K index).
// ISA layout: lane holds N column = lane%16; lane<16 -> K={0..7,16..23}, lane>=16 -> K={8..15,24..31}.
__device__ __forceinline__ v16h load_frag_b(const _Float16* __restrict__ m,
                                            int k0, int n0, int ld, int lane) {
  const _Float16* p = m + (size_t)(k0 + ((lane & 16) ? 8 : 0)) * ld + n0 + (lane & 15);
  v16h b;
#pragma unroll
  for (int i = 0; i < 8; ++i) { b[i] = p[(size_t)i * ld]; b[i + 8] = p[(size_t)(16 + i) * ld]; }
  return b;
}

// B fragment straight out of a contiguous 32x64 LDS tile (row stride 64 halfs).
__device__ __forceinline__ v16h load_frag_b16(const _Float16* p) {
  v16h b;
#pragma unroll
  for (int i = 0; i < 8; ++i) { b[i] = p[i * 64]; b[i + 8] = p[(16 + i) * 64]; }
  return b;
}

// ---------- elementwise prep kernels ----------

__global__ void f32_to_f16_kernel(const float* __restrict__ in,
                                  _Float16* __restrict__ out, int n) {
  int i = blockIdx.x * blockDim.x + threadIdx.x;
  if (i < n) out[i] = (_Float16)in[i];
}

// w: [N][K] row-major f32  ->  wt: [K][N] row-major f16
__global__ void transpose_f32_to_f16_kernel(const float* __restrict__ w,
                                            _Float16* __restrict__ wt,
                                            int N, int K) {
  int i = blockIdx.x * blockDim.x + threadIdx.x;
  if (i >= N * K) return;
  int n = i / K, k = i % K;
  wt[(size_t)k * N + n] = (_Float16)w[i];
}

// ---------- fused GEMM with TDM weight staging ----------
// out[M][N] = A[M][K] @ Bt[K][N] + bias
// block = 128 threads = 4 waves stacked along M; each wave does 16(M) x 64(N).
// The 32(K) x 64(N) weight tile is DMAed into LDS once per k-chunk by the
// Tensor Data Mover (one issue from wave 0, TENSORcnt-tracked), then all four
// waves read their B fragments from LDS (strided 16-bit LDS reads) -> 4x less global
// traffic on the weight stream.

template <typename OutT>
__global__ void wmma_gemm_kernel(const _Float16* __restrict__ A,
                                 const _Float16* __restrict__ Bt,
                                 const float* __restrict__ bias,
                                 OutT* __restrict__ out,
                                 int M, int N, int K) {
  __shared__ _Float16 lds_b[32 * 64];  // 4 KB K-major weight tile
  const int lane = threadIdx.x & 31;
  const int wave = threadIdx.x >> 5;
  const int m0 = blockIdx.y * 64 + wave * 16;
  const int n0 = blockIdx.x * 64;

  const unsigned int lds_off = (unsigned int)(uintptr_t)(&lds_b[0]);

  v8f c0 = {}, c1 = {}, c2 = {}, c3 = {};
  for (int k0 = 0; k0 < K; k0 += 32) {
    if (wave == 0) {
      // --- Tensor DMA descriptor (D#), CDNA5 ISA ---
      unsigned long long gaddr =
          (unsigned long long)(uintptr_t)(Bt + (size_t)k0 * N + n0);
      v4u g0;
      g0[0] = 1u;                                  // count=1, user, no gather
      g0[1] = lds_off;                             // lds_addr
      g0[2] = (unsigned int)gaddr;                 // global_addr[31:0]
      g0[3] = (unsigned int)((gaddr >> 32) & 0x01FFFFFFull) |
              0x80000000u;                         // global_addr[56:32] | type=2
      v8u g1;
      g1[0] = 0x00010000u;                         // data_size=1 (2 bytes)
      g1[1] = ((unsigned int)N & 0xFFFFu) << 16;   // tensor_dim0[15:0]
      g1[2] = (((unsigned int)N >> 16) & 0xFFFFu) |
              (((unsigned int)K & 0xFFFFu) << 16); // tensor_dim0[31:16] | tensor_dim1[15:0]
      g1[3] = (((unsigned int)K >> 16) & 0xFFFFu) |
              (64u << 16);                         // tensor_dim1[31:16] | tile_dim0=64
      g1[4] = 32u;                                 // tile_dim1=32, tile_dim2=0
      g1[5] = (unsigned int)N;                     // tensor_dim0_stride[31:0]
      g1[6] = 0u;                                  // stride hi | dim1_stride lo
      g1[7] = 0u;
      v4u gz = {0u, 0u, 0u, 0u};                   // groups 2/3: 2-D tile, unused
      asm volatile("tensor_load_to_lds %0, %1, %2, %3"
                   :: "s"(g0), "s"(g1), "s"(gz), "s"(gz)
                   : "memory");
      __builtin_amdgcn_s_wait_tensorcnt(0);
    }
    __syncthreads();  // publish LDS weight tile to all 4 waves

    v16h a = load_frag_a(A, m0, k0, K, lane);
    if (k0 + 32 < K)  // global_prefetch_b8 on the activation stream
      __builtin_prefetch(A + (size_t)(m0 + (lane & 15)) * K + k0 + 32, 0, 1);

    const _Float16* lb = &lds_b[((lane & 16) ? 8 : 0) * 64 + (lane & 15)];
    v16h b0 = load_frag_b16(lb + 0);
    v16h b1 = load_frag_b16(lb + 16);
    v16h b2 = load_frag_b16(lb + 32);
    v16h b3 = load_frag_b16(lb + 48);
    c0 = wmma16(a, b0, c0);
    c1 = wmma16(a, b1, c1);
    c2 = wmma16(a, b2, c2);
    c3 = wmma16(a, b3, c3);
    __syncthreads();  // all waves done reading before next DMA overwrites
  }

  const int mb = m0 + ((lane & 16) ? 8 : 0);
  const int nl = lane & 15;
#pragma unroll
  for (int i = 0; i < 8; ++i) {
    size_t row = (size_t)(mb + i) * N;
    out[row + n0 + 0  + nl] = (OutT)(c0[i] + bias[n0 + 0  + nl]);
    out[row + n0 + 16 + nl] = (OutT)(c1[i] + bias[n0 + 16 + nl]);
    out[row + n0 + 32 + nl] = (OutT)(c2[i] + bias[n0 + 32 + nl]);
    out[row + n0 + 48 + nl] = (OutT)(c3[i] + bias[n0 + 48 + nl]);
  }
}

// ---------- RoPE + head split: qkv[M][3C] -> q[BH][T][D], kT[BH][D][T], v[BH][T][D] ----------

__global__ void rope_split_kernel(const _Float16* __restrict__ qkv,
                                  _Float16* __restrict__ q,
                                  _Float16* __restrict__ kT,
                                  _Float16* __restrict__ v) {
  int idx = blockIdx.x * blockDim.x + threadIdx.x;
  if (idx >= BH_ * T_ * 32) return;
  int j  = idx & 31;              // rotation pair index, 0..31
  int t  = (idx >> 5) % T_;
  int bh = idx / (32 * T_);
  int b = bh >> 4, h = bh & 15;

  float inv = __expf(-((float)(2 * j)) * (9.210340371976184f / 64.0f)); // 10000^(-2j/D)
  float ang = (float)t * inv;
  float sn, cs;
  __sincosf(ang, &sn, &cs);

  size_t rowbase = ((size_t)b * T_ + t) * N3C;
  size_t qi = rowbase + (size_t)h * D_;
  size_t ki = qi + C_;
  size_t vi = qi + 2 * C_;
  size_t od = ((size_t)bh * T_ + t) * D_;

  float xe = (float)qkv[qi + 2 * j], xo = (float)qkv[qi + 2 * j + 1];
  q[od + j]      = (_Float16)(xe * cs - xo * sn);
  q[od + j + 32] = (_Float16)(xe * sn + xo * cs);

  xe = (float)qkv[ki + 2 * j]; xo = (float)qkv[ki + 2 * j + 1];
  size_t kb = (size_t)bh * D_ * T_;
  kT[kb + (size_t)j * T_ + t]        = (_Float16)(xe * cs - xo * sn);
  kT[kb + (size_t)(j + 32) * T_ + t] = (_Float16)(xe * sn + xo * cs);

  v[od + 2 * j]     = qkv[vi + 2 * j];
  v[od + 2 * j + 1] = qkv[vi + 2 * j + 1];
}

// ---------- causal flash attention: one wave per 16-row query tile ----------
// y (f16) written in [B][T][H][D] layout == [M][C] for the projection GEMM.

__global__ void flash_attn_kernel(const _Float16* __restrict__ q,
                                  const _Float16* __restrict__ kT,
                                  const _Float16* __restrict__ v,
                                  _Float16* __restrict__ y) {
  __shared__ _Float16 plds[16 * 32];   // P tile staging: C-layout -> A-layout
  const int lane = threadIdx.x & 31;
  const int t0 = blockIdx.x * 16;
  const int bh = blockIdx.y;
  const int b = bh >> 4, h = bh & 15;

  const _Float16* qb = q  + (size_t)bh * T_ * D_;
  const _Float16* kb = kT + (size_t)bh * D_ * T_;
  const _Float16* vb = v  + (size_t)bh * T_ * D_;

  v16h aq0 = load_frag_a(qb, t0, 0,  D_, lane);   // Q d=0..31
  v16h aq1 = load_frag_a(qb, t0, 32, D_, lane);   // Q d=32..63

  const int mb = (lane & 16) ? 8 : 0;  // C-layout row offset for this lane half
  const int nl = lane & 15;            // C-layout column within tile

  float mrow[8], lrow[8], alpha[8], p0[8], p1[8];
#pragma unroll
  for (int i = 0; i < 8; ++i) { mrow[i] = -1e30f; lrow[i] = 0.0f; }
  v8f acc0 = {}, acc1 = {}, acc2 = {}, acc3 = {};
  const float scale = 0.125f;  // 1/sqrt(64)

  for (int s0 = 0; s0 < t0 + 16; s0 += 32) {
    // S = Q @ K^T : two 16x16 score tiles (cols s0..s0+15, s0+16..s0+31)
    v16h b00 = load_frag_b(kb, 0,  s0,      T_, lane);
    v16h b01 = load_frag_b(kb, 32, s0,      T_, lane);
    v16h b10 = load_frag_b(kb, 0,  s0 + 16, T_, lane);
    v16h b11 = load_frag_b(kb, 32, s0 + 16, T_, lane);
    v8f sf0 = {}, sf1 = {};
    sf0 = wmma16(aq0, b00, sf0);
    sf0 = wmma16(aq1, b01, sf0);
    sf1 = wmma16(aq0, b10, sf1);
    sf1 = wmma16(aq1, b11, sf1);

    // online softmax, per row (row stats live across the 16 lanes of a half)
#pragma unroll
    for (int i = 0; i < 8; ++i) {
      int row = t0 + mb + i;
      float v0 = sf0[i] * scale; if (s0 + nl      > row) v0 = -1e30f;
      float v1 = sf1[i] * scale; if (s0 + 16 + nl > row) v1 = -1e30f;
      float rm = fmaxf(v0, v1);
      rm = fmaxf(rm, __shfl_xor(rm, 1, 32));
      rm = fmaxf(rm, __shfl_xor(rm, 2, 32));
      rm = fmaxf(rm, __shfl_xor(rm, 4, 32));
      rm = fmaxf(rm, __shfl_xor(rm, 8, 32));
      float mnew = fmaxf(mrow[i], rm);
      alpha[i] = __expf(mrow[i] - mnew);
      mrow[i] = mnew;
      p0[i] = __expf(v0 - mnew);
      p1[i] = __expf(v1 - mnew);
      float rs = p0[i] + p1[i];
      rs += __shfl_xor(rs, 1, 32);
      rs += __shfl_xor(rs, 2, 32);
      rs += __shfl_xor(rs, 4, 32);
      rs += __shfl_xor(rs, 8, 32);
      lrow[i] = lrow[i] * alpha[i] + rs;
    }
#pragma unroll
    for (int i = 0; i < 8; ++i) {
      acc0[i] *= alpha[i]; acc1[i] *= alpha[i];
      acc2[i] *= alpha[i]; acc3[i] *= alpha[i];
    }

    // C-layout P -> LDS -> A-layout P (cross-lane reshape within the wave)
#pragma unroll
    for (int i = 0; i < 8; ++i) {
      plds[(mb + i) * 32 + nl]      = (_Float16)p0[i];
      plds[(mb + i) * 32 + nl + 16] = (_Float16)p1[i];
    }
    asm volatile("s_wait_dscnt 0x0" ::: "memory");
    v16h ap;
#pragma unroll
    for (int i = 0; i < 8; ++i) {
      ap[i]     = plds[(lane & 15) * 32 + mb + i];
      ap[i + 8] = plds[(lane & 15) * 32 + mb + 16 + i];
    }
    asm volatile("s_wait_dscnt 0x0" ::: "memory");

    // y += P @ V  (four 16-wide d tiles)
    v16h bv0 = load_frag_b(vb, s0, 0,  D_, lane);
    v16h bv1 = load_frag_b(vb, s0, 16, D_, lane);
    v16h bv2 = load_frag_b(vb, s0, 32, D_, lane);
    v16h bv3 = load_frag_b(vb, s0, 48, D_, lane);
    acc0 = wmma16(ap, bv0, acc0);
    acc1 = wmma16(ap, bv1, acc1);
    acc2 = wmma16(ap, bv2, acc2);
    acc3 = wmma16(ap, bv3, acc3);
  }

#pragma unroll
  for (int i = 0; i < 8; ++i) {
    int trow = t0 + mb + i;
    float inv = 1.0f / lrow[i];
    size_t base = (((size_t)b * T_ + trow) * H_ + h) * D_;
    y[base + 0  + nl] = (_Float16)(acc0[i] * inv);
    y[base + 16 + nl] = (_Float16)(acc1[i] * inv);
    y[base + 32 + nl] = (_Float16)(acc2[i] * inv);
    y[base + 48 + nl] = (_Float16)(acc3[i] * inv);
  }
}

// ---------- launch ----------

extern "C" void kernel_launch(void* const* d_in, const int* in_sizes, int n_in,
                              void* d_out, int out_size, void* d_ws, size_t ws_size,
                              hipStream_t stream) {
  (void)in_sizes; (void)n_in; (void)out_size; (void)ws_size;
  const float* x      = (const float*)d_in[0];
  const float* w_attn = (const float*)d_in[1];
  const float* b_attn = (const float*)d_in[2];
  const float* w_proj = (const float*)d_in[3];
  const float* b_proj = (const float*)d_in[4];
  float* out = (float*)d_out;

  char* ws = (char*)d_ws;
  _Float16* x16   = (_Float16*)ws; ws += (size_t)M_ * C_   * 2;  //  8 MB
  _Float16* waT   = (_Float16*)ws; ws += (size_t)C_ * N3C  * 2;  //  6 MB
  _Float16* wpT   = (_Float16*)ws; ws += (size_t)C_ * C_   * 2;  //  2 MB
  _Float16* qkv16 = (_Float16*)ws; ws += (size_t)M_ * N3C  * 2;  // 24 MB
  _Float16* q16   = (_Float16*)ws; ws += (size_t)BH_ * T_ * D_ * 2;  // 8 MB
  _Float16* kT16  = (_Float16*)ws; ws += (size_t)BH_ * D_ * T_ * 2;  // 8 MB
  _Float16* v16   = (_Float16*)ws; ws += (size_t)BH_ * T_ * D_ * 2;  // 8 MB
  _Float16* y16   = (_Float16*)ws; ws += (size_t)M_ * C_ * 2;        // 8 MB

  // 1) precision conversion / weight transposition
  f32_to_f16_kernel<<<(M_ * C_ + 255) / 256, 256, 0, stream>>>(x, x16, M_ * C_);
  transpose_f32_to_f16_kernel<<<(N3C * C_ + 255) / 256, 256, 0, stream>>>(w_attn, waT, N3C, C_);
  transpose_f32_to_f16_kernel<<<(C_ * C_ + 255) / 256, 256, 0, stream>>>(w_proj, wpT, C_, C_);

  // 2) QKV projection: [4096 x 1024] @ [1024 x 3072] + b
  wmma_gemm_kernel<_Float16><<<dim3(N3C / 64, M_ / 64), 128, 0, stream>>>(
      x16, waT, b_attn, qkv16, M_, N3C, C_);

  // 3) RoPE + split into q / k^T / v
  rope_split_kernel<<<(BH_ * T_ * 32 + 255) / 256, 256, 0, stream>>>(qkv16, q16, kT16, v16);

  // 4) causal flash attention (one wave per 16-row query tile)
  flash_attn_kernel<<<dim3(T_ / 16, BH_), 32, 0, stream>>>(q16, kT16, v16, y16);

  // 5) output projection: [4096 x 1024] @ [1024 x 1024] + b -> f32 out
  wmma_gemm_kernel<float><<<dim3(C_ / 64, M_ / 64), 128, 0, stream>>>(
      y16, wpT, b_proj, out, M_, C_, C_);
}
